// MultiHeadAttention_52896817218049
// MI455X (gfx1250) — matrix operand order repaired and checked by reading, not compiled, not run on hardware
//
#include <hip/hip_runtime.h>
#include <hip/hip_bf16.h>

typedef __bf16 bf16;
typedef __attribute__((ext_vector_type(16))) __bf16 bf16x16;
typedef __attribute__((ext_vector_type(8)))  __bf16 bf16x8;
typedef __attribute__((ext_vector_type(8)))  float  f32x8;

#define DEV __device__ __forceinline__

DEV f32x8 zero8() {
  f32x8 z;
#pragma unroll
  for (int e = 0; e < 8; ++e) z[e] = 0.0f;
  return z;
}

// ---- WMMA fragment loaders (CDNA5 16-bit layouts, wave32) ----
// A matrix (M=16 x K=32): lane r (0..15) holds K=0..7 (elems 0..7) and
// K=16..23 (elems 8..15); lane r+16 holds K=8..15 and K=24..31.
DEV bf16x16 load_frag_a(const bf16* __restrict__ base, int ld) {
  const int lane = threadIdx.x & 31;
  const int r = lane & 15, hi = lane >> 4;
  const bf16* p = base + (size_t)r * ld + hi * 8;
  bf16x8 lo = *(const bf16x8*)(p);
  bf16x8 hv = *(const bf16x8*)(p + 16);
  bf16x16 f;
#pragma unroll
  for (int i = 0; i < 8; ++i) { f[i] = lo[i]; f[i + 8] = hv[i]; }
  return f;
}

// B matrix (K=32 x N=16), source stored as B^T (N x K) row-major:
// lane n (0..15) holds K=0..7 and K=16..23; lane n+16 holds K=8..15 and K=24..31.
DEV bf16x16 load_frag_b(const bf16* __restrict__ base, int ld) {
  const int lane = threadIdx.x & 31;
  const int n = lane & 15, hi = lane >> 4;
  const bf16* p = base + (size_t)n * ld + hi * 8;
  bf16x8 lo = *(const bf16x8*)(p);
  bf16x8 hv = *(const bf16x8*)(p + 16);
  bf16x16 f;
#pragma unroll
  for (int i = 0; i < 8; ++i) { f[i] = lo[i]; f[i + 8] = hv[i]; }
  return f;
}

DEV f32x8 wmma_bf16(bf16x16 a, bf16x16 b, f32x8 c) {
  return __builtin_amdgcn_wmma_f32_16x16x32_bf16(false, a, false, b, (short)0, c,
                                                 false, false);
}

// ---------------- fp32 -> bf16 cast ----------------
__global__ void cast_f32_to_bf16(const float* __restrict__ x, bf16* __restrict__ y,
                                 int n) {
  int i = blockIdx.x * blockDim.x + threadIdx.x;
  const int stride = gridDim.x * blockDim.x;
  for (; i < n; i += stride) y[i] = (bf16)x[i];
}

// ---------------- GEMM: Y = A (MxK) * Bw^T (Bw is NxK) + bias ----------------
// block = 256 threads = 8 waves (2x4), block tile 128x256, wave tile 64x64.
template <bool OUT_F32>
__global__ __launch_bounds__(256) void gemm_wmma(const bf16* __restrict__ A,
                                                 const bf16* __restrict__ Bw,
                                                 const float* __restrict__ bias,
                                                 void* __restrict__ Yv, int M, int N,
                                                 int K) {
  const int wave = threadIdx.x >> 5;
  const int lane = threadIdx.x & 31;
  const int wm = wave & 1, wn = wave >> 1;
  const int m0 = blockIdx.y * 128 + wm * 64;
  const int n0 = blockIdx.x * 256 + wn * 64;

  f32x8 acc[4][4];
#pragma unroll
  for (int i = 0; i < 4; ++i)
#pragma unroll
    for (int j = 0; j < 4; ++j) acc[i][j] = zero8();

  for (int k0 = 0; k0 < K; k0 += 32) {
    if (k0 + 32 < K) {
      __builtin_prefetch(A + (size_t)m0 * K + k0 + 32, 0, 3);
      __builtin_prefetch(Bw + (size_t)n0 * K + k0 + 32, 0, 3);
    }
    bf16x16 af[4], bfr[4];
#pragma unroll
    for (int i = 0; i < 4; ++i)
      af[i] = load_frag_a(A + (size_t)(m0 + 16 * i) * K + k0, K);
#pragma unroll
    for (int j = 0; j < 4; ++j)
      bfr[j] = load_frag_b(Bw + (size_t)(n0 + 16 * j) * K + k0, K);
#pragma unroll
    for (int i = 0; i < 4; ++i)
#pragma unroll
      for (int j = 0; j < 4; ++j) acc[i][j] = wmma_bf16(af[i], bfr[j], acc[i][j]);
  }

  const int col = lane & 15, hi = lane >> 4;
#pragma unroll
  for (int j = 0; j < 4; ++j) {
    const int n = n0 + 16 * j + col;
    const float bv = bias[n];
#pragma unroll
    for (int i = 0; i < 4; ++i)
#pragma unroll
      for (int e = 0; e < 8; ++e) {
        const int row = m0 + 16 * i + e + 8 * hi;
        const float v = acc[i][j][e] + bv;
        if (OUT_F32)
          ((float*)Yv)[(size_t)row * N + n] = v;
        else
          ((bf16*)Yv)[(size_t)row * N + n] = (bf16)v;
      }
  }
}

// ---------------- Flash attention ----------------
// grid = (S/256, H, B); block = 256 (8 waves); wave owns 32 q-rows.
__global__ __launch_bounds__(256) void attn_wmma(const bf16* __restrict__ Q,
                                                 const bf16* __restrict__ Km,
                                                 const bf16* __restrict__ V,
                                                 bf16* __restrict__ Ctx) {
  constexpr int Sq = 2048, Dm = 1024, DK = 64;
  __shared__ __align__(16) bf16 ldsK[32][72];      // K chunk, row = kv idx, col = d
  __shared__ __align__(16) bf16 ldsVT[64][40];     // V^T chunk, row = d, col = kv idx
  __shared__ __align__(16) bf16 ldsP[8][32][40];   // per-wave P (32q x 32k)

  const int b = blockIdx.z, h = blockIdx.y, qb = blockIdx.x;
  const int wave = threadIdx.x >> 5;
  const int lane = threadIdx.x & 31;
  const int col = lane & 15, hi = lane >> 4;
  const int rowQ0 = b * Sq + qb * 256 + wave * 32;
  const int colH = h * DK;

  // Preload Q fragments for this wave's 32 rows (2 m-tiles x 2 k-steps).
  bf16x16 aq[2][2];
#pragma unroll
  for (int t = 0; t < 2; ++t)
#pragma unroll
    for (int s = 0; s < 2; ++s)
      aq[t][s] = load_frag_a(Q + (size_t)(rowQ0 + 16 * t) * Dm + colH + 32 * s, Dm);

  f32x8 o[2][4], mst[2], lst[2];
#pragma unroll
  for (int t = 0; t < 2; ++t) {
#pragma unroll
    for (int jv = 0; jv < 4; ++jv) o[t][jv] = zero8();
#pragma unroll
    for (int e = 0; e < 8; ++e) { mst[t][e] = -1.0e30f; lst[t][e] = 0.0f; }
  }

  const int tRow = threadIdx.x >> 3;       // 0..31 (kv row within chunk)
  const int tSeg = (threadIdx.x & 7) * 8;  // 0..56 (d segment)
  const float c2 = 0.125f * 1.44269504088896340736f;  // 1/sqrt(64) * log2(e)

  for (int kc = 0; kc < Sq; kc += 32) {
    __syncthreads();
    {  // cooperative staging of K chunk and transposed V chunk
      const size_t grow = (size_t)(b * Sq + kc + tRow) * Dm + colH + tSeg;
      bf16x8 kv = *(const bf16x8*)(Km + grow);
      *(bf16x8*)&ldsK[tRow][tSeg] = kv;
      bf16x8 vv = *(const bf16x8*)(V + grow);
#pragma unroll
      for (int e = 0; e < 8; ++e) ldsVT[tSeg + e][tRow] = vv[e];
    }
    __syncthreads();

    // scores S = Q * K^T (2 m-tiles x 2 n-tiles, K-dim = 64 = 2 steps)
    bf16x16 bk[2][2];
#pragma unroll
    for (int j = 0; j < 2; ++j)
#pragma unroll
      for (int s = 0; s < 2; ++s) bk[j][s] = load_frag_b(&ldsK[16 * j][32 * s], 72);

    f32x8 sc[2][2];
#pragma unroll
    for (int t = 0; t < 2; ++t)
#pragma unroll
      for (int j = 0; j < 2; ++j) {
        f32x8 z = zero8();
        z = wmma_bf16(aq[t][0], bk[j][0], z);
        z = wmma_bf16(aq[t][1], bk[j][1], z);
        sc[t][j] = z;
      }

    // online softmax (rows live across 16-lane halves -> width-16 reductions)
#pragma unroll
    for (int t = 0; t < 2; ++t) {
      f32x8 mx;
#pragma unroll
      for (int e = 0; e < 8; ++e) {
        sc[t][0][e] *= c2;
        sc[t][1][e] *= c2;
        mx[e] = fmaxf(sc[t][0][e], sc[t][1][e]);
      }
#pragma unroll
      for (int off = 8; off >= 1; off >>= 1)
#pragma unroll
        for (int e = 0; e < 8; ++e) mx[e] = fmaxf(mx[e], __shfl_xor(mx[e], off, 16));

      f32x8 mnew, corr, rsum;
#pragma unroll
      for (int e = 0; e < 8; ++e) {
        mnew[e] = fmaxf(mst[t][e], mx[e]);
        corr[e] = exp2f(mst[t][e] - mnew[e]);
        rsum[e] = 0.0f;
      }
#pragma unroll
      for (int j = 0; j < 2; ++j)
#pragma unroll
        for (int e = 0; e < 8; ++e) {
          const float p = exp2f(sc[t][j][e] - mnew[e]);
          rsum[e] += p;
          ldsP[wave][16 * t + e + 8 * hi][16 * j + col] = (bf16)p;  // C -> A relayout
        }
#pragma unroll
      for (int off = 8; off >= 1; off >>= 1)
#pragma unroll
        for (int e = 0; e < 8; ++e) rsum[e] += __shfl_xor(rsum[e], off, 16);
#pragma unroll
      for (int e = 0; e < 8; ++e) {
        lst[t][e] = lst[t][e] * corr[e] + rsum[e];
        mst[t][e] = mnew[e];
      }
#pragma unroll
      for (int jv = 0; jv < 4; ++jv)
#pragma unroll
        for (int e = 0; e < 8; ++e) o[t][jv][e] *= corr[e];
    }

    // O += P * V  (A = P from LDS relayout, B^T = V^T from LDS)
    bf16x16 bv[4];
#pragma unroll
    for (int jv = 0; jv < 4; ++jv) bv[jv] = load_frag_b(&ldsVT[16 * jv][0], 40);
#pragma unroll
    for (int t = 0; t < 2; ++t) {
      bf16x16 ap = load_frag_a(&ldsP[wave][16 * t][0], 40);
#pragma unroll
      for (int jv = 0; jv < 4; ++jv) o[t][jv] = wmma_bf16(ap, bv[jv], o[t][jv]);
    }
  }

  // finalize: divide by row sums, write context (b,s,h,dk) == (row, D) layout
#pragma unroll
  for (int t = 0; t < 2; ++t)
#pragma unroll
    for (int e = 0; e < 8; ++e) {
      const float inv = 1.0f / lst[t][e];
      const int row = rowQ0 + 16 * t + e + 8 * hi;
#pragma unroll
      for (int jv = 0; jv < 4; ++jv)
        Ctx[(size_t)row * Dm + colH + 16 * jv + col] = (bf16)(o[t][jv][e] * inv);
    }
}

// ---------------- launcher ----------------
extern "C" void kernel_launch(void* const* d_in, const int* in_sizes, int n_in,
                              void* d_out, int out_size, void* d_ws, size_t ws_size,
                              hipStream_t stream) {
  (void)in_sizes; (void)n_in; (void)out_size; (void)ws_size;
  const float* X  = (const float*)d_in[0];
  // d_in[1] = attention_mask (unused by the reference forward)
  const float* Wq = (const float*)d_in[2];
  const float* bq = (const float*)d_in[3];
  const float* Wk = (const float*)d_in[4];
  const float* bk = (const float*)d_in[5];
  const float* Wv = (const float*)d_in[6];
  const float* bv = (const float*)d_in[7];
  const float* Wo = (const float*)d_in[8];
  const float* bo = (const float*)d_in[9];

  constexpr int NROW = 4 * 2048;  // B*S
  constexpr int D = 1024;

  char* ws = (char*)d_ws;
  bf16* Xh  = (bf16*)ws; ws += (size_t)NROW * D * 2;
  bf16* Wqh = (bf16*)ws; ws += (size_t)D * D * 2;
  bf16* Wkh = (bf16*)ws; ws += (size_t)D * D * 2;
  bf16* Wvh = (bf16*)ws; ws += (size_t)D * D * 2;
  bf16* Woh = (bf16*)ws; ws += (size_t)D * D * 2;
  bf16* Qh  = (bf16*)ws; ws += (size_t)NROW * D * 2;
  bf16* Kh  = (bf16*)ws; ws += (size_t)NROW * D * 2;
  bf16* Vh  = (bf16*)ws; ws += (size_t)NROW * D * 2;
  bf16* Ch  = (bf16*)ws; ws += (size_t)NROW * D * 2;

  cast_f32_to_bf16<<<4096, 256, 0, stream>>>(X, Xh, NROW * D);
  cast_f32_to_bf16<<<2048, 256, 0, stream>>>(Wq, Wqh, D * D);
  cast_f32_to_bf16<<<2048, 256, 0, stream>>>(Wk, Wkh, D * D);
  cast_f32_to_bf16<<<2048, 256, 0, stream>>>(Wv, Wvh, D * D);
  cast_f32_to_bf16<<<2048, 256, 0, stream>>>(Wo, Woh, D * D);

  dim3 ggrid(D / 256, NROW / 128);  // (4, 64)
  gemm_wmma<false><<<ggrid, 256, 0, stream>>>(Xh, Wqh, bq, Qh, NROW, D, D);
  gemm_wmma<false><<<ggrid, 256, 0, stream>>>(Xh, Wkh, bk, Kh, NROW, D, D);
  gemm_wmma<false><<<ggrid, 256, 0, stream>>>(Xh, Wvh, bv, Vh, NROW, D, D);

  dim3 agrid(2048 / 256, 16, 4);  // (q-blocks, H, B)
  attn_wmma<<<agrid, 256, 0, stream>>>(Qh, Kh, Vh, Ch);

  gemm_wmma<true><<<ggrid, 256, 0, stream>>>(Ch, Woh, bo, d_out, NROW, D, D);
}
